// AANet_54657753809024
// MI455X (gfx1250) — compile-verified, matrix operand order checked by reading, not run on hardware
//
#include <hip/hip_runtime.h>
#include <hip/hip_bf16.h>

typedef __attribute__((ext_vector_type(16))) _Float16 v16h;
typedef __attribute__((ext_vector_type(8)))  float    v8f;

// ---------------------------------------------------------------------------
// device helpers
// ---------------------------------------------------------------------------
static __device__ __forceinline__ float leakyf(float x, float s) {
    return x >= 0.f ? x : s * x;
}
static __device__ __forceinline__ float sigmoidf(float x) {
    return 1.0f / (1.0f + expf(-x));
}

// One 16x16x32 f16->f32 WMMA from LDS tiles.
// sA: 16x32 (row major, stride lda halves), sB: 32x16 (row major, stride ldb).
// Lane layout per CDNA5 ISA 7.12.2 (16-bit A 16x32; B mirrored).
static __device__ __forceinline__ void wmma_tile(const _Float16* sA, int lda,
                                                 const _Float16* sB, int ldb,
                                                 v8f& acc, int lane) {
    v16h a, b;
    const int m  = lane & 15;
    const int hi = (lane >> 4) & 1;
#pragma unroll
    for (int i = 0; i < 16; ++i) {
        const int j  = i >> 1;
        const int p  = i & 1;
        const int kk = ((j < 4) ? (2 * j) : (16 + 2 * (j - 4))) + (hi ? 8 : 0) + p;
        a[i] = sA[m * lda + kk];
        b[i] = sB[kk * ldb + m];
    }
    acc = __builtin_amdgcn_wmma_f32_16x16x32_f16(false, a, false, b,
                                                 (short)0, acc, false, false);
}

// ---------------------------------------------------------------------------
// implicit-GEMM convolution, NCHW, OIHW weights, f32 I/O, f16 WMMA math
// 64x64 output tile / workgroup, 8 waves; each wave owns a 16x32 strip
// (2 accumulators), K staged 64-deep in LDS -> 4 v_wmma per stage.
// ---------------------------------------------------------------------------
__global__ void conv_wmma_kernel(const float* __restrict__ src,
                                 const float* __restrict__ wgt,
                                 const float* __restrict__ bias,
                                 float* __restrict__ dst,
                                 int Nb, int Cin, int H, int W, int Cout,
                                 int OH, int OW, int stride, int pad, int dil,
                                 int KH, int KW, int dstCtot, int dstCoff) {
    const int M    = Nb * OH * OW;
    const int Ktot = Cin * KH * KW;
    const int tileM = blockIdx.x * 64;
    const int tileN = blockIdx.y * 64;

    __shared__ _Float16 sA[64][72];   // M x K
    __shared__ _Float16 sB[64][72];   // K x N

    const int tid  = threadIdx.x;     // 0..255
    const int lane = tid & 31;
    const int wave = tid >> 5;        // 0..7  -> 4(M) x 2(N) wave grid
    const int wm   = (wave >> 1) * 16;
    const int wn   = (wave & 1) * 32;

    v8f acc0 = {};
    v8f acc1 = {};

    for (int k0 = 0; k0 < Ktot; k0 += 64) {
#pragma unroll
        for (int i = 0; i < 16; ++i) {
            const int e  = tid + i * 256;     // 0..4095
            const int mr = e >> 6, kk = e & 63;
            // ---- A: input patch element (pixel mr, reduction kk)
            const int gm = tileM + mr;
            const int gk = k0 + kk;
            float va = 0.f;
            if (gm < M && gk < Ktot) {
                const int ow  = gm % OW;
                const int tmp = gm / OW;
                const int oh  = tmp % OH;
                const int nb  = tmp / OH;
                const int kx  = gk % KW;
                const int t2  = gk / KW;
                const int ky  = t2 % KH;
                const int ci  = t2 / KH;
                const int iy  = oh * stride - pad + ky * dil;
                const int ix  = ow * stride - pad + kx * dil;
                if (iy >= 0 && iy < H && ix >= 0 && ix < W)
                    va = src[(((long long)nb * Cin + ci) * H + iy) * W + ix];
            }
            sA[mr][kk] = (_Float16)va;
            // ---- B: weight element (reduction mr, out-channel kk)
            const int gn  = tileN + kk;
            const int gk2 = k0 + mr;
            float vb = 0.f;
            if (gn < Cout && gk2 < Ktot) {
                vb = wgt[(long long)gn * Ktot + gk2];
                if (gk2 + 64 < Ktot)
                    __builtin_prefetch(&wgt[(long long)gn * Ktot + gk2 + 64], 0, 1);
            }
            sB[mr][kk] = (_Float16)vb;
        }
        __syncthreads();
#pragma unroll
        for (int kc = 0; kc < 64; kc += 32) {
            wmma_tile(&sA[wm][kc], 72, &sB[kc][wn],      72, acc0, lane);
            wmma_tile(&sA[wm][kc], 72, &sB[kc][wn + 16], 72, acc1, lane);
        }
        __syncthreads();
    }

    const int nn  = lane & 15;
    const int co0 = tileN + wn + nn;
    const int co1 = co0 + 16;
#pragma unroll
    for (int r = 0; r < 8; ++r) {
        const int m_ = r + ((lane >> 4) << 3);
        const int gm = tileM + wm + m_;
        if (gm < M) {
            const int ow  = gm % OW;
            const int tmp = gm / OW;
            const int oh  = tmp % OH;
            const int nb  = tmp / OH;
            const long long base = ((long long)nb * dstCtot + dstCoff) * OH;
            if (co0 < Cout)
                dst[((base + (long long)co0 * OH) + oh) * OW + ow] = acc0[r] + bias[co0];
            if (co1 < Cout)
                dst[((base + (long long)co1 * OH) + oh) * OW + ow] = acc1[r] + bias[co1];
        }
    }
}

// ---------------------------------------------------------------------------
// generic GEMM: C[M,N] = A[M,K] * B[K,N]  (+bias, optional relu)
// same 64x64 / 8-wave / K=64-stage tiling as the conv kernel
// ---------------------------------------------------------------------------
__global__ void gemm_wmma_kernel(const float* __restrict__ A,
                                 const float* __restrict__ B,
                                 const float* __restrict__ bias,
                                 float* __restrict__ C,
                                 int M, int Nn, int K, int relu) {
    const int tileM = blockIdx.x * 64;
    const int tileN = blockIdx.y * 64;
    __shared__ _Float16 sA[64][72];
    __shared__ _Float16 sB[64][72];
    const int tid  = threadIdx.x;
    const int lane = tid & 31;
    const int wave = tid >> 5;
    const int wm   = (wave >> 1) * 16;
    const int wn   = (wave & 1) * 32;
    v8f acc0 = {};
    v8f acc1 = {};

    for (int k0 = 0; k0 < K; k0 += 64) {
#pragma unroll
        for (int i = 0; i < 16; ++i) {
            const int e  = tid + i * 256;
            const int r_ = e >> 6, c_ = e & 63;
            const int gm = tileM + r_, gk = k0 + c_;
            sA[r_][c_] = (_Float16)((gm < M && gk < K) ? A[(long long)gm * K + gk] : 0.f);
            const int gn = tileN + c_, gk2 = k0 + r_;
            float vb = 0.f;
            if (gn < Nn && gk2 < K) {
                vb = B[(long long)gk2 * Nn + gn];
                if (gk2 + 64 < K) __builtin_prefetch(&B[(long long)(gk2 + 64) * Nn + gn], 0, 1);
            }
            sB[r_][c_] = (_Float16)vb;
        }
        __syncthreads();
#pragma unroll
        for (int kc = 0; kc < 64; kc += 32) {
            wmma_tile(&sA[wm][kc], 72, &sB[kc][wn],      72, acc0, lane);
            wmma_tile(&sA[wm][kc], 72, &sB[kc][wn + 16], 72, acc1, lane);
        }
        __syncthreads();
    }

    const int nn  = lane & 15;
    const int gn0 = tileN + wn + nn;
    const int gn1 = gn0 + 16;
#pragma unroll
    for (int r = 0; r < 8; ++r) {
        const int gm = tileM + wm + r + ((lane >> 4) << 3);
        if (gm < M) {
            if (gn0 < Nn) {
                float v = acc0[r] + (bias ? bias[gn0] : 0.f);
                if (relu) v = fmaxf(v, 0.f);
                C[(long long)gm * Nn + gn0] = v;
            }
            if (gn1 < Nn) {
                float v = acc1[r] + (bias ? bias[gn1] : 0.f);
                if (relu) v = fmaxf(v, 0.f);
                C[(long long)gm * Nn + gn1] = v;
            }
        }
    }
}

// ---------------------------------------------------------------------------
// device-to-device copy through LDS using the CDNA5 async DMA path
// (global_load_async_to_lds / global_store_async_from_lds, ASYNCcnt)
// n must be a multiple of 1024 floats; every lane moves 16B per stage.
// ---------------------------------------------------------------------------
__global__ void async_copy_kernel(const float* __restrict__ src,
                                  float* __restrict__ dst, long long n) {
    __shared__ __align__(16) float buf[1024];
    const int tid = threadIdx.x;                       // 256 threads
    const unsigned lds = (unsigned)(unsigned long long)(uintptr_t)&buf[tid * 4];
    for (long long base = (long long)blockIdx.x * 1024; base < n;
         base += (long long)gridDim.x * 1024) {
        const float* s = src + base + tid * 4;
        float* d = dst + base + tid * 4;
        asm volatile("global_load_async_to_lds_b128 %0, %1, off"
                     :: "v"(lds), "v"(s) : "memory");
        asm volatile("s_wait_asynccnt 0" ::: "memory");
        asm volatile("global_store_async_from_lds_b128 %0, %1, off"
                     :: "v"(d), "v"(lds) : "memory");
        asm volatile("s_wait_asynccnt 0" ::: "memory");
    }
}

// ---------------------------------------------------------------------------
// KV einsum: for each (n,h): KV[d,v] = sum_s Ke[n,s,h,d] * V[n,s,h,v]/S
// single wave per (n,h); D=16, WMMA over K=S
// ---------------------------------------------------------------------------
__global__ void kv_wmma_kernel(const float* __restrict__ Ke,
                               const float* __restrict__ V,
                               float* __restrict__ KV,
                               int S, int Hh) {
    const int nh = blockIdx.x;
    const int h  = nh % Hh;
    const int n  = nh / Hh;
    __shared__ _Float16 sA[16][40];
    __shared__ _Float16 sB[32][24];
    const int lane = threadIdx.x;
    const float invS = 1.0f / (float)S;
    v8f acc = {};

    for (int s0 = 0; s0 < S; s0 += 32) {
#pragma unroll
        for (int i = 0; i < 16; ++i) {
            const int e  = lane * 16 + i;       // 0..511
            const int d  = e >> 5, ss = e & 31; // A element (d, ss)
            sA[d][ss] = (_Float16)Ke[(((long long)n * S + (s0 + ss)) * Hh + h) * 16 + d];
            const int s2 = e >> 4, vv = e & 15; // B element (s2, vv)
            sB[s2][vv] = (_Float16)(V[(((long long)n * S + (s0 + s2)) * Hh + h) * 16 + vv] * invS);
        }
        __syncthreads();
        wmma_tile(&sA[0][0], 40, &sB[0][0], 24, acc, lane);
        __syncthreads();
    }
    const int nn = lane & 15;
#pragma unroll
    for (int r = 0; r < 8; ++r) {
        const int m_ = r + ((lane >> 4) << 3);
        KV[((long long)nh * 16 + m_) * 16 + nn] = acc[r];
    }
}

// K.sum over s: KS[n,h,d]
__global__ void ksum_kernel(const float* __restrict__ Ke, float* __restrict__ KS,
                            int S, int Hh) {
    const int o = blockIdx.x;
    const int d = o & 15;
    const int h = (o >> 4) % Hh;
    const int n = o / (16 * Hh);
    __shared__ float red[256];
    float s = 0.f;
    for (int ss = threadIdx.x; ss < S; ss += 256)
        s += Ke[(((long long)n * S + ss) * Hh + h) * 16 + d];
    red[threadIdx.x] = s;
    __syncthreads();
    for (int st = 128; st > 0; st >>= 1) {
        if (threadIdx.x < st) red[threadIdx.x] += red[threadIdx.x + st];
        __syncthreads();
    }
    if (threadIdx.x == 0) KS[o] = red[0];
}

// out[n,l,h,v] = sum_d Qe*KV * Z * S ; Z = 1/(Qe.KS + eps)
__global__ void attn_out_kernel(const float* __restrict__ Qe,
                                const float* __restrict__ KV,
                                const float* __restrict__ KS,
                                float* __restrict__ Out, int L, int Hh, long long tot) {
    const long long idx = (long long)blockIdx.x * blockDim.x + threadIdx.x;
    if (idx >= tot) return;
    const int h = (int)(idx % Hh);
    const long long nl = idx / Hh;
    const int l = (int)(nl % L);
    const int n = (int)(nl / L);
    const float* q  = &Qe[(((long long)n * L + l) * Hh + h) * 16];
    const float* kv = &KV[((long long)n * Hh + h) * 256];
    const float* ks = &KS[((long long)n * Hh + h) * 16];
    float qr[16];
    float den = 1e-6f;
#pragma unroll
    for (int d = 0; d < 16; ++d) { qr[d] = q[d]; den += qr[d] * ks[d]; }
    const float zs = (1.0f / den) * (float)L;
    float* o = &Out[(((long long)n * L + l) * Hh + h) * 16];
#pragma unroll
    for (int v = 0; v < 16; ++v) {
        float s = 0.f;
#pragma unroll
        for (int d = 0; d < 16; ++d) s += qr[d] * kv[d * 16 + v];
        o[v] = s * zs;
    }
}

// ---------------------------------------------------------------------------
// normalization / activation kernels
// ---------------------------------------------------------------------------
__global__ void inorm_stats_kernel(const float* __restrict__ X, float* __restrict__ mv,
                                   long long P, float slope) {
    const int pl = blockIdx.x;
    const float* p = X + (long long)pl * P;
    __shared__ float r1[256], r2[256];
    float s = 0.f, ss = 0.f;
    for (long long i = threadIdx.x; i < P; i += 256) {
        const float v = leakyf(p[i], slope);
        s += v; ss += v * v;
    }
    r1[threadIdx.x] = s; r2[threadIdx.x] = ss;
    __syncthreads();
    for (int st = 128; st > 0; st >>= 1) {
        if (threadIdx.x < st) { r1[threadIdx.x] += r1[threadIdx.x + st];
                                r2[threadIdx.x] += r2[threadIdx.x + st]; }
        __syncthreads();
    }
    if (threadIdx.x == 0) {
        const float mean = r1[0] / (float)P;
        mv[pl * 2]     = mean;
        mv[pl * 2 + 1] = r2[0] / (float)P - mean * mean;
    }
}

__global__ void inorm_apply_kernel(float* __restrict__ X, const float* __restrict__ mv,
                                   long long P, long long total, float slope) {
    for (long long i = (long long)blockIdx.x * blockDim.x + threadIdx.x; i < total;
         i += (long long)gridDim.x * blockDim.x) {
        const int pl = (int)(i / P);
        const float m = mv[pl * 2], v = mv[pl * 2 + 1];
        X[i] = (leakyf(X[i], slope) - m) * rsqrtf(v + 1e-5f);
    }
}

__global__ void prelu_kernel(float* __restrict__ X, const float* __restrict__ a,
                             long long total) {
    const float aa = a[0];
    for (long long i = (long long)blockIdx.x * blockDim.x + threadIdx.x; i < total;
         i += (long long)gridDim.x * blockDim.x) {
        const float x = X[i];
        X[i] = x >= 0.f ? x : aa * x;
    }
}

__global__ void elu1_kernel(float* __restrict__ X, long long total) {
    for (long long i = (long long)blockIdx.x * blockDim.x + threadIdx.x; i < total;
         i += (long long)gridDim.x * blockDim.x) {
        const float x = X[i];
        X[i] = x >= 0.f ? x + 1.0f : expf(x);   // elu(x)+1
    }
}

__global__ void layernorm_kernel(const float* __restrict__ X, const float* __restrict__ Res,
                                 const float* __restrict__ g, const float* __restrict__ b,
                                 float* __restrict__ Out, int C) {
    __shared__ float red[256];
    const int row = blockIdx.x;
    const int c   = threadIdx.x;
    const float v = X[(long long)row * C + c];
    red[c] = v;
    __syncthreads();
    for (int s = C >> 1; s > 0; s >>= 1) {
        if (c < s) red[c] += red[c + s];
        __syncthreads();
    }
    const float mean = red[0] / (float)C;
    __syncthreads();
    const float dv = v - mean;
    red[c] = dv * dv;
    __syncthreads();
    for (int s = C >> 1; s > 0; s >>= 1) {
        if (c < s) red[c] += red[c + s];
        __syncthreads();
    }
    const float var = red[0] / (float)C;
    const float y = dv * rsqrtf(var + 1e-5f) * g[c] + b[c];
    Out[(long long)row * C + c] = Res ? Res[(long long)row * C + c] + y : y;
}

// ---------------------------------------------------------------------------
// simam kernels (pool pyramid 8/4/2/1, max-upsample, variance, sigmoid gate)
// ---------------------------------------------------------------------------
__global__ void simam_pool8_kernel(const float* __restrict__ X, float* __restrict__ P4,
                                   int H, int W) {
    const int pl   = blockIdx.x;
    const int cell = threadIdx.x; // 0..63
    const int bh = H / 8, bw = W / 8;
    const int cy = cell >> 3, cx = cell & 7;
    const float* p = X + (long long)pl * H * W;
    float s = 0.f;
    for (int yy = 0; yy < bh; ++yy)
        for (int xx = 0; xx < bw; ++xx)
            s += p[(cy * bh + yy) * W + cx * bw + xx];
    P4[pl * 64 + cell] = s / (float)(bh * bw);
}

__global__ void simam_down_kernel(const float* __restrict__ P4, float* __restrict__ P3,
                                  float* __restrict__ P2, float* __restrict__ P1, int nc) {
    const int pl = blockIdx.x * blockDim.x + threadIdx.x;
    if (pl >= nc) return;
    float m3[16], m2[4];
    for (int y = 0; y < 4; ++y)
        for (int x = 0; x < 4; ++x) {
            const float v = (P4[pl * 64 + (2 * y) * 8 + 2 * x] +
                             P4[pl * 64 + (2 * y) * 8 + 2 * x + 1] +
                             P4[pl * 64 + (2 * y + 1) * 8 + 2 * x] +
                             P4[pl * 64 + (2 * y + 1) * 8 + 2 * x + 1]) * 0.25f;
            m3[y * 4 + x] = v;
            P3[pl * 16 + y * 4 + x] = v;
        }
    for (int y = 0; y < 2; ++y)
        for (int x = 0; x < 2; ++x) {
            const float v = (m3[(2 * y) * 4 + 2 * x] + m3[(2 * y) * 4 + 2 * x + 1] +
                             m3[(2 * y + 1) * 4 + 2 * x] + m3[(2 * y + 1) * 4 + 2 * x + 1]) * 0.25f;
            m2[y * 2 + x] = v;
            P2[pl * 4 + y * 2 + x] = v;
        }
    P1[pl] = (m2[0] + m2[1] + m2[2] + m2[3]) * 0.25f;
}

static __device__ __forceinline__ float simam_mm(const float* P4, const float* P3,
                                                 const float* P2, const float* P1,
                                                 int pl, int y, int x, int H, int W) {
    const float m4 = P4[pl * 64 + (y / (H / 8)) * 8 + (x / (W / 8))];
    const float m3 = P3[pl * 16 + (y / (H / 4)) * 4 + (x / (W / 4))];
    const float m2 = P2[pl * 4 + (y / (H / 2)) * 2 + (x / (W / 2))];
    const float m1 = P1[pl];
    return fmaxf(fmaxf(m1, m2), fmaxf(m3, m4));
}

__global__ void simam_vsum_kernel(const float* __restrict__ X, const float* __restrict__ P4,
                                  const float* __restrict__ P3, const float* __restrict__ P2,
                                  const float* __restrict__ P1, float* __restrict__ VV,
                                  int H, int W) {
    const int pl = blockIdx.x;
    const long long P = (long long)H * W;
    const float* p = X + (long long)pl * P;
    __shared__ float red[256];
    float s = 0.f;
    for (long long i = threadIdx.x; i < P; i += 256) {
        const int y = (int)(i / W), x = (int)(i % W);
        const float d = p[i] - simam_mm(P4, P3, P2, P1, pl, y, x, H, W);
        s += d * d;
    }
    red[threadIdx.x] = s;
    __syncthreads();
    for (int st = 128; st > 0; st >>= 1) {
        if (threadIdx.x < st) red[threadIdx.x] += red[threadIdx.x + st];
        __syncthreads();
    }
    if (threadIdx.x == 0) VV[pl] = red[0] / (float)(H * W - 64);
}

__global__ void simam_apply_kernel(float* __restrict__ X, const float* __restrict__ P4,
                                   const float* __restrict__ P3, const float* __restrict__ P2,
                                   const float* __restrict__ P1, const float* __restrict__ VV,
                                   int H, int W, long long total) {
    const long long P = (long long)H * W;
    for (long long i = (long long)blockIdx.x * blockDim.x + threadIdx.x; i < total;
         i += (long long)gridDim.x * blockDim.x) {
        const int pl = (int)(i / P);
        const long long r = i % P;
        const int y = (int)(r / W), x = (int)(r % W);
        const float d = X[i] - simam_mm(P4, P3, P2, P1, pl, y, x, H, W);
        const float t = (d * d) / (4.0f * (VV[pl] + 1e-4f)) + 0.5f;
        X[i] *= sigmoidf(t);
    }
}

// ---------------------------------------------------------------------------
// layout / misc kernels
// ---------------------------------------------------------------------------
// T[n, l, c] = X3[n, c, y, x] + pos_encoding(c, y, x)  ; C=128, 64x64
__global__ void pe_transpose_kernel(const float* __restrict__ X3, float* __restrict__ T,
                                    long long total) {
    const long long idx = (long long)blockIdx.x * blockDim.x + threadIdx.x;
    if (idx >= total) return;
    const int c = (int)(idx % 128);
    const long long rem = idx / 128;
    const int l = (int)(rem % 4096);
    const int n = (int)(rem / 4096);
    const int y = l >> 6, x = l & 63;
    const int j = c >> 2, r = c & 3;
    const float dv  = expf(-2.0f * (float)j);
    const float pos = (r < 2) ? (float)(x + 1) : (float)(y + 1);
    const float pe  = (r & 1) ? cosf(dv * pos) : sinf(dv * pos);
    T[idx] = X3[(((long long)n * 128 + c) * 64 + y) * 64 + x] + pe;
}

// G[n, c, y, x] = T[n, y*64+x, c]
__global__ void untranspose_kernel(const float* __restrict__ T, float* __restrict__ G,
                                   long long total) {
    const long long idx = (long long)blockIdx.x * blockDim.x + threadIdx.x;
    if (idx >= total) return;
    const int x = (int)(idx & 63);
    const int y = (int)((idx >> 6) & 63);
    const int c = (int)((idx >> 12) & 127);
    const int n = (int)(idx >> 19);
    G[idx] = T[((long long)n * 4096 + (y * 64 + x)) * 128 + c];
}

__global__ void bilinear_kernel(const float* __restrict__ X, float* __restrict__ O,
                                int h, int w, int OH, int OW, long long total) {
    for (long long i = (long long)blockIdx.x * blockDim.x + threadIdx.x; i < total;
         i += (long long)gridDim.x * blockDim.x) {
        const int ox = (int)(i % OW);
        const int oy = (int)((i / OW) % OH);
        const int pc = (int)(i / ((long long)OW * OH));
        const float fy = (OH > 1) ? (float)oy * (float)(h - 1) / (float)(OH - 1) : 0.f;
        const float fx = (OW > 1) ? (float)ox * (float)(w - 1) / (float)(OW - 1) : 0.f;
        const int y0 = (int)floorf(fy); const int y1 = min(y0 + 1, h - 1);
        const int x0 = (int)floorf(fx); const int x1 = min(x0 + 1, w - 1);
        const float wy = fy - (float)y0, wx = fx - (float)x0;
        const float* p = X + (long long)pc * h * w;
        const float v = (p[y0 * w + x0] * (1.f - wx) + p[y0 * w + x1] * wx) * (1.f - wy) +
                        (p[y1 * w + x0] * (1.f - wx) + p[y1 * w + x1] * wx) * wy;
        O[i] = v;
    }
}

__global__ void cat2_kernel(const float* __restrict__ A, const float* __restrict__ B,
                            float* __restrict__ D, int C, long long total) {
    for (long long i = (long long)blockIdx.x * blockDim.x + threadIdx.x; i < total;
         i += (long long)gridDim.x * blockDim.x) {
        const int c2 = (int)(i % (2 * C));
        const long long m = i / (2 * C);
        D[i] = (c2 < C) ? A[m * C + c2] : B[m * C + (c2 - C)];
    }
}

// ---------------------------------------------------------------------------
// host orchestration
// ---------------------------------------------------------------------------
static inline int conv_out(int H, int pad, int dil, int k, int s) {
    return (H + 2 * pad - dil * (k - 1) - 1) / s + 1;
}

static void run_conv(hipStream_t st, const float* src, const float* w, const float* b,
                     float* dst, int Nb, int Cin, int H, int W, int Cout,
                     int stride, int pad, int dil, int K, int dstCtot, int dstCoff) {
    const int OH = conv_out(H, pad, dil, K, stride);
    const int OW = conv_out(W, pad, dil, K, stride);
    const int M = Nb * OH * OW;
    dim3 g((M + 63) / 64, (Cout + 63) / 64);
    conv_wmma_kernel<<<g, 256, 0, st>>>(src, w, b, dst, Nb, Cin, H, W, Cout, OH, OW,
                                        stride, pad, dil, K, K, dstCtot, dstCoff);
}

static void run_gemm(hipStream_t st, const float* A, const float* B, const float* bias,
                     float* C, int M, int N, int K, int relu) {
    dim3 g((M + 63) / 64, (N + 63) / 64);
    gemm_wmma_kernel<<<g, 256, 0, st>>>(A, B, bias, C, M, N, K, relu);
}

static void run_copy(hipStream_t st, const float* s, float* d, long long n) {
    async_copy_kernel<<<1024, 256, 0, st>>>(s, d, n);   // n % 1024 == 0
}

extern "C" void kernel_launch(void* const* d_in, const int* in_sizes, int n_in,
                              void* d_out, int out_size, void* d_ws, size_t ws_size,
                              hipStream_t stream) {
    (void)out_size; (void)ws_size;
    // locate x by its unique element count (2*3*512*512)
    int xi = 0;
    for (int i = 0; i < n_in; ++i)
        if (in_sizes[i] == 2 * 3 * 512 * 512) { xi = i; break; }
    const float* X = (const float*)d_in[xi];
    const int pb = (xi == 0) ? 1 : 0;
#define PP(i) ((const float*)d_in[pb + (i)])
    // params flattened (sorted keys, jax pytree order), 0-based:
    // b0:0-5 b1:6-11 b2:12-17 b3:18-23  (b1,b2,b3,w1,w2,w3 each)
    // merge:24-28 (a,b1,b2,w1,w2)  s1:29-33  s2:34-38 (a,b1,b2,w1,w2)
    // s3:39-40 (b,w)
    // t1:41-50 t2:51-60 (ln1_b,ln1_s,ln2_b,ln2_s,mw1,mw2,wk,wm,wq,wv)

    float* ws = (float*)d_ws;
    size_t off = 0;
    auto carve = [&](size_t n) { float* p = ws + off; off += n; return p; };
    float* A   = carve(33554432);
    float* Bf  = carve(33554432);
    float* X1  = carve(8388608);
    float* X2  = carve(2097152);
    float* X3  = carve(1048576);
    float* F   = carve(6291456);
    float* G2  = carve(4194304);
    float* T   = carve(1048576);
    float* Qb  = carve(1048576);
    float* Kb  = carve(1048576);
    float* Vb  = carve(1048576);
    float* AO  = carve(1048576);
    float* MSG = carve(1048576);
    float* CAT = carve(2097152);
    float* H1  = carve(2097152);
    float* H2  = carve(1048576);
    float* KVb = carve(4096);
    float* KS  = carve(256);
    float* MV  = carve(2048);
    float* P4  = carve(24576);
    float* P3  = carve(6144);
    float* P2  = carve(1536);
    float* P1  = carve(384);
    float* VV  = carve(384);

    const int Nb = 2;
    auto inorm = [&](float* Xp, int nc, long long P, float slope) {
        inorm_stats_kernel<<<nc, 256, 0, stream>>>(Xp, MV, P, slope);
        const long long tot = (long long)nc * P;
        inorm_apply_kernel<<<(unsigned)((tot + 255) / 256), 256, 0, stream>>>(Xp, MV, P, tot, slope);
    };
    auto simam = [&](float* Xp, int nc, int Hs, int Ws) {
        simam_pool8_kernel<<<nc, 64, 0, stream>>>(Xp, P4, Hs, Ws);
        simam_down_kernel<<<(nc + 63) / 64, 64, 0, stream>>>(P4, P3, P2, P1, nc);
        simam_vsum_kernel<<<nc, 256, 0, stream>>>(Xp, P4, P3, P2, P1, VV, Hs, Ws);
        const long long tot = (long long)nc * Hs * Ws;
        simam_apply_kernel<<<(unsigned)((tot + 255) / 256), 256, 0, stream>>>(
            Xp, P4, P3, P2, P1, VV, Hs, Ws, tot);
    };

    // ---------------- b0 (512x512) ----------------
    run_conv(stream, X,  PP(3), PP(0), A,  Nb, 3, 512, 512, 32, 1, 1, 1, 3, 32, 0);
    inorm(A, Nb * 32, 512LL * 512, 0.1f);
    run_conv(stream, A,  PP(4), PP(1), Bf, Nb, 32, 512, 512, 64, 1, 2, 2, 3, 64, 0);
    inorm(Bf, Nb * 64, 512LL * 512, 0.1f);
    run_conv(stream, Bf, PP(5), PP(2), A,  Nb, 64, 512, 512, 64, 1, 1, 1, 3, 64, 0);
    inorm(A, Nb * 64, 512LL * 512, 0.1f);                       // x0 in A

    // ---------------- b1 (-> 256x256) ----------------
    run_conv(stream, A,  PP(9),  PP(6), Bf, Nb, 64, 512, 512, 128, 2, 1, 1, 4, 128, 0);
    inorm(Bf, Nb * 128, 256LL * 256, 0.1f);
    run_conv(stream, Bf, PP(10), PP(7), A,  Nb, 128, 256, 256, 64, 1, 2, 2, 3, 64, 0);
    inorm(A, Nb * 64, 256LL * 256, 0.1f);
    run_conv(stream, A,  PP(11), PP(8), Bf, Nb, 64, 256, 256, 64, 1, 1, 1, 3, 64, 0);
    inorm(Bf, Nb * 64, 256LL * 256, 0.1f);                      // x1 in Bf
    run_copy(stream, Bf, X1, 8388608);

    // ---------------- b2 (-> 128x128) ----------------
    run_conv(stream, X1, PP(15), PP(12), A,  Nb, 64, 256, 256, 128, 2, 1, 1, 4, 128, 0);
    inorm(A, Nb * 128, 128LL * 128, 0.1f);
    run_conv(stream, A,  PP(16), PP(13), Bf, Nb, 128, 128, 128, 64, 1, 2, 2, 3, 64, 0);
    inorm(Bf, Nb * 64, 128LL * 128, 0.1f);
    run_conv(stream, Bf, PP(17), PP(14), A,  Nb, 64, 128, 128, 64, 1, 1, 1, 3, 64, 0);
    inorm(A, Nb * 64, 128LL * 128, 0.1f);                       // x2 in A
    run_copy(stream, A, X2, 2097152);

    // ---------------- b3 (-> 64x64, final_na=False) ----------------
    run_conv(stream, X2, PP(21), PP(18), A,  Nb, 64, 128, 128, 128, 2, 1, 1, 4, 128, 0);
    inorm(A, Nb * 128, 64LL * 64, 0.1f);
    run_conv(stream, A,  PP(22), PP(19), Bf, Nb, 128, 64, 64, 128, 1, 2, 2, 3, 128, 0);
    inorm(Bf, Nb * 128, 64LL * 64, 0.2f);
    run_conv(stream, Bf, PP(23), PP(20), A,  Nb, 128, 64, 64, 128, 1, 1, 1, 3, 128, 0);
    run_copy(stream, A, X3, 1048576);                            // x3

    // ---------------- skip1: x1 -> F[:, 0:64] ----------------
    run_conv(stream, X1, PP(32), PP(30), A, Nb, 64, 256, 256, 128, 2, 1, 1, 4, 128, 0);
    simam(A, Nb * 128, 128, 128);
    prelu_kernel<<<16384, 256, 0, stream>>>(A, PP(29), (long long)Nb * 128 * 128 * 128);
    inorm(A, Nb * 128, 128LL * 128, 1.0f);
    run_conv(stream, A, PP(33), PP(31), F, Nb, 128, 128, 128, 64, 1, 1, 1, 3, 192, 0);

    // ---------------- skip2: x2 -> F[:, 64:128] ----------------
    run_conv(stream, X2, PP(37), PP(35), A, Nb, 64, 128, 128, 128, 1, 1, 1, 3, 128, 0);
    simam(A, Nb * 128, 128, 128);
    prelu_kernel<<<16384, 256, 0, stream>>>(A, PP(34), (long long)Nb * 128 * 128 * 128);
    inorm(A, Nb * 128, 128LL * 128, 1.0f);
    run_conv(stream, A, PP(38), PP(36), F, Nb, 128, 128, 128, 64, 1, 1, 1, 3, 192, 64);

    // ---------------- transformer on x3 ----------------
    const long long Ttot = (long long)Nb * 128 * 4096;
    pe_transpose_kernel<<<(unsigned)((Ttot + 255) / 256), 256, 0, stream>>>(X3, T, Ttot);

    const int M = Nb * 4096, C = 128, Hh = 8, S = 4096;
    for (int layer = 0; layer < 2; ++layer) {
        const int t = 41 + layer * 10; // t1 @41, t2 @51
        const float *ln1b = PP(t), *ln1s = PP(t + 1), *ln2b = PP(t + 2), *ln2s = PP(t + 3);
        const float *mw1 = PP(t + 4), *mw2 = PP(t + 5);
        const float *wk = PP(t + 6), *wm = PP(t + 7), *wq = PP(t + 8), *wv = PP(t + 9);
        run_gemm(stream, T, wq, nullptr, Qb, M, C, C, 0);
        run_gemm(stream, T, wk, nullptr, Kb, M, C, C, 0);
        run_gemm(stream, T, wv, nullptr, Vb, M, C, C, 0);
        elu1_kernel<<<4096, 256, 0, stream>>>(Qb, (long long)M * C);
        elu1_kernel<<<4096, 256, 0, stream>>>(Kb, (long long)M * C);
        kv_wmma_kernel<<<Nb * Hh, 32, 0, stream>>>(Kb, Vb, KVb, S, Hh);
        ksum_kernel<<<Nb * Hh * 16, 256, 0, stream>>>(Kb, KS, S, Hh);
        const long long atot = (long long)Nb * 4096 * Hh;
        attn_out_kernel<<<(unsigned)((atot + 255) / 256), 256, 0, stream>>>(Qb, KVb, KS, AO, 4096, Hh, atot);
        run_gemm(stream, AO, wm, nullptr, MSG, M, C, C, 0);
        layernorm_kernel<<<M, C, 0, stream>>>(MSG, nullptr, ln1s, ln1b, MSG, C);
        const long long ctot = (long long)M * 2 * C;
        cat2_kernel<<<(unsigned)((ctot + 255) / 256), 256, 0, stream>>>(T, MSG, CAT, C, ctot);
        run_gemm(stream, CAT, mw1, nullptr, H1, M, 2 * C, 2 * C, 1);
        run_gemm(stream, H1, mw2, nullptr, H2, M, C, 2 * C, 0);
        layernorm_kernel<<<M, C, 0, stream>>>(H2, T, ln2s, ln2b, T, C);  // T = T + ln(H2)
    }

    // ---------------- x3s: upsample + conv -> F[:, 128:192] ----------------
    untranspose_kernel<<<(unsigned)((Ttot + 255) / 256), 256, 0, stream>>>(T, X3, Ttot);
    const long long btot = (long long)Nb * 128 * 128 * 128;
    bilinear_kernel<<<(unsigned)((btot + 255) / 256), 256, 0, stream>>>(X3, G2, 64, 64, 128, 128, btot);
    run_conv(stream, G2, PP(40), PP(39), F, Nb, 128, 128, 128, 64, 1, 1, 1, 3, 192, 128);
    // bilinear to (128,128) afterwards is identity -> skipped

    // ---------------- merge ----------------
    simam(F, Nb * 192, 128, 128);
    prelu_kernel<<<24576, 256, 0, stream>>>(F, PP(24), (long long)Nb * 192 * 128 * 128);
    inorm(F, Nb * 192, 128LL * 128, 1.0f);
    run_conv(stream, F, PP(27), PP(25), A, Nb, 192, 128, 128, 128, 1, 1, 1, 3, 128, 0);
    inorm(A, Nb * 128, 128LL * 128, 1.0f);
    run_conv(stream, A, PP(28), PP(26), (float*)d_out, Nb, 128, 128, 128, 128, 1, 1, 1, 3, 128, 0);
#undef PP
}
